// CPC_Net_71725953843809
// MI455X (gfx1250) — matrix operand order repaired and checked by reading, not compiled, not run on hardware
//
#include <hip/hip_runtime.h>
#include <hip/hip_bf16.h>

typedef __attribute__((ext_vector_type(16))) _Float16 v16h;
typedef __attribute__((ext_vector_type(8)))  float    v8f;

// ---------------- problem sizes ----------------
#define B_SZ  16      // batch
#define NP    16      // num prediction steps / windows
#define NB    10      // negatives
#define NWIN  192     // NP*NB (xb) + NP (xc) + NP (xp)
#define T_IN  3000
#define CCH   2
#define F1    16      // conv channels
#define KW    50      // conv kernel length
#define TO1   2951    // 3000-50+1
#define PG1   227     // 2951/13 (exact)
#define TO2   178     // 227-50+1
#define PG2   13      // 178//13 (169 used)
#define NH    100
#define FEATN 416     // 16*13*2
#define EPSBN 1e-5f

// ---------------- workspace layout (floats) ----------------
// requires ws_size >= ~92 MiB
static constexpr size_t OFF_S1SUM = 0;
static constexpr size_t OFF_S1SQ  = (size_t)NWIN * F1;
static constexpr size_t OFF_S2SUM = 2 * (size_t)NWIN * F1;
static constexpr size_t OFF_S2SQ  = 3 * (size_t)NWIN * F1;
static constexpr size_t OFF_P1    = 4 * (size_t)NWIN * F1;                  // 12288
static constexpr size_t SZ_P1     = (size_t)NWIN * B_SZ * F1 * PG1 * CCH;   // 22,315,008
static constexpr size_t OFF_P2    = OFF_P1 + SZ_P1;
static constexpr size_t SZ_P2     = (size_t)NWIN * B_SZ * F1 * PG2 * CCH;   // 1,277,952
static constexpr size_t OFF_FEAT  = OFF_P2 + SZ_P2;
static constexpr size_t SZ_FEAT   = (size_t)NWIN * B_SZ * NH;               // 307,200
static constexpr size_t OFF_HN    = OFF_FEAT + SZ_FEAT;
static constexpr size_t WS_FLOATS = OFF_HN + (size_t)B_SZ * NH;

// window w -> base pointer of x[b, :, :] (T_IN x CCH, row-major)
__device__ __forceinline__ const float* win_base(const float* Xb, const float* Xc,
                                                 const float* Xp, int w, int b) {
  if (w < NP * NB) {
    int p = w / NB, n = w % NB;
    return Xb + ((size_t)((b * NP + p) * NB + n)) * T_IN * CCH;
  } else if (w < NP * NB + NP) {
    int s = w - NP * NB;
    return Xc + ((size_t)(b * NP + s)) * T_IN * CCH;
  } else {
    int p = w - NP * NB - NP;
    return Xp + ((size_t)(b * NP + p)) * T_IN * CCH;
  }
}

// A-fragment K base, contiguous K order (k = 16*hi + 2*r), vgpr r, lane-half hi
__device__ __forceinline__ int a_kbase(int r, int hi) {
  return 2 * r + 16 * hi;
}

// ================= K1: spatial-mix + conv2 (WMMA) + ReLU + BN-stats + pool-max =================
__global__ __launch_bounds__(256) void k1_conv2(
    const float* __restrict__ Xc, const float* __restrict__ Xp, const float* __restrict__ Xb,
    const float* __restrict__ w1, const float* __restrict__ b1,
    const float* __restrict__ w2, const float* __restrict__ b2,
    float* __restrict__ ws) {
  __shared__ _Float16 aW[F1][64];          // w2 zero-padded K50->64, f16
  __shared__ int      pmax[CCH][F1][PG1];  // pooled relu-max (float bits, >=0)
  __shared__ float    ssum[F1], ssq[F1];

  const int w = blockIdx.x / B_SZ, b = blockIdx.x % B_SZ;
  const int tid = threadIdx.x, lane = tid & 31, hi = lane >> 4, ln = lane & 15, wv = tid >> 5;

  for (int i = tid; i < F1 * 64; i += 256) {
    int f = i >> 6, k = i & 63;
    aW[f][k] = (_Float16)(k < KW ? w2[f * KW + k] : 0.0f);
  }
  for (int i = tid; i < CCH * F1 * PG1; i += 256) (&pmax[0][0][0])[i] = 0;
  if (tid < F1) { ssum[tid] = 0.f; ssq[tid] = 0.f; }
  __syncthreads();

  // hoist A fragments (w2, K chunks 0..31 and 32..63)
  v16h a0, a1;
#pragma unroll
  for (int r = 0; r < 8; ++r) {
    int kb = a_kbase(r, hi);
    a0[2 * r] = aW[ln][kb];      a0[2 * r + 1] = aW[ln][kb + 1];
    a1[2 * r] = aW[ln][32 + kb]; a1[2 * r + 1] = aW[ln][32 + kb + 1];
  }
  float b2r[8];
#pragma unroll
  for (int v = 0; v < 8; ++v) b2r[v] = b2[v + 8 * hi];

  const float* xw = win_base(Xb, Xc, Xp, w, b);
  float lsum[8] = {0, 0, 0, 0, 0, 0, 0, 0}, lsq[8] = {0, 0, 0, 0, 0, 0, 0, 0};

  const int NT = (TO1 + 15) / 16;  // 185 tiles of 16 outputs
  for (int job = wv; job < CCH * NT; job += 8) {
    int c = job / NT, t0 = (job % NT) * 16;
    float wc0 = w1[c * CCH + 0], wc1 = w1[c * CCH + 1], bc = b1[c];
    // B fragments: row K = 16*hi + h (+32 for 2nd), col N = ln; value y1[t0+N+K]
    v16h bf0, bf1;
#pragma unroll
    for (int h = 0; h < 16; ++h) {
      int t = t0 + ln + 16 * hi + h;
      float y = 0.f, y2 = 0.f;
      if (t < T_IN) y = xw[t * CCH] * wc0 + xw[t * CCH + 1] * wc1 + bc;
      int t2 = t + 32;
      if (t2 < T_IN) y2 = xw[t2 * CCH] * wc0 + xw[t2 * CCH + 1] * wc1 + bc;
      bf0[h] = (_Float16)y;
      bf1[h] = (_Float16)y2;
    }
    v8f acc;
#pragma unroll
    for (int v = 0; v < 8; ++v) acc[v] = b2r[v];  // bias in accumulator
    acc = __builtin_amdgcn_wmma_f32_16x16x32_f16(false, a0, false, bf0, (short)0, acc, false, false);
    acc = __builtin_amdgcn_wmma_f32_16x16x32_f16(false, a1, false, bf1, (short)0, acc, false, false);

    int tO = t0 + ln;
    if (tO < TO1) {
      int g = tO / 13;
#pragma unroll
      for (int v = 0; v < 8; ++v) {
        float r = fmaxf(acc[v], 0.f);
        lsum[v] += r; lsq[v] += r * r;
        atomicMax(&pmax[c][v + 8 * hi][g], __float_as_int(r));
      }
    }
  }
#pragma unroll
  for (int v = 0; v < 8; ++v) {
    atomicAdd(&ssum[v + 8 * hi], lsum[v]);
    atomicAdd(&ssq[v + 8 * hi], lsq[v]);
  }
  __syncthreads();
  if (tid < F1) {
    atomicAdd(&ws[OFF_S1SUM + (size_t)w * F1 + tid], ssum[tid]);
    atomicAdd(&ws[OFF_S1SQ  + (size_t)w * F1 + tid], ssq[tid]);
  }
  float* p1 = ws + OFF_P1 + ((size_t)(w * B_SZ + b)) * F1 * PG1 * CCH;
  for (int i = tid; i < F1 * PG1 * CCH; i += 256) {
    int c = i % CCH, g = (i / CCH) % PG1, f = i / (CCH * PG1);
    p1[((size_t)f * PG1 + g) * CCH + c] = __int_as_float(pmax[c][f][g]);
  }
}

// ================= K2: BN1-fold + conv3 (WMMA, K=800=25x32) + ReLU + BN-stats + pool-max =================
__global__ __launch_bounds__(256) void k2_conv3(
    const float* __restrict__ w3, const float* __restrict__ b3,
    const float* __restrict__ g1, const float* __restrict__ be1,
    float* __restrict__ ws) {
  __shared__ _Float16 aW[F1][800];         // w3[f][g*50+kk] f16 (25.6 KB)
  __shared__ _Float16 in3[CCH][F1][PG1];   // BN1-applied pooled1 (14.5 KB)
  __shared__ int      pmax[CCH][F1][PG2];
  __shared__ float    ssum[F1], ssq[F1], scl[F1], shf[F1];

  const int w = blockIdx.x / B_SZ, b = blockIdx.x % B_SZ;
  const int tid = threadIdx.x, lane = tid & 31, hi = lane >> 4, ln = lane & 15, wv = tid >> 5;

  if (tid < F1) {
    const float N1 = (float)(B_SZ * TO1 * CCH);
    float m = ws[OFF_S1SUM + (size_t)w * F1 + tid] / N1;
    float var = ws[OFF_S1SQ + (size_t)w * F1 + tid] / N1 - m * m;
    float s = g1[tid] * rsqrtf(var + EPSBN);
    scl[tid] = s; shf[tid] = be1[tid] - m * s;
    ssum[tid] = 0.f; ssq[tid] = 0.f;
  }
  for (int i = tid; i < F1 * 800; i += 256) aW[i / 800][i % 800] = (_Float16)w3[i];
  for (int i = tid; i < CCH * F1 * PG2; i += 256) (&pmax[0][0][0])[i] = 0;
  __syncthreads();

  const float* p1 = ws + OFF_P1 + ((size_t)(w * B_SZ + b)) * F1 * PG1 * CCH;
  for (int i = tid; i < CCH * F1 * PG1; i += 256) {
    int f = i / (PG1 * CCH), r = i % (PG1 * CCH), t = r / CCH, c = r % CCH;
    in3[c][f][t] = (_Float16)(p1[((size_t)f * PG1 + t) * CCH + c] * scl[f] + shf[f]);
  }
  __syncthreads();

  float b3r[8];
#pragma unroll
  for (int v = 0; v < 8; ++v) b3r[v] = b3[v + 8 * hi];
  float lsum[8] = {0, 0, 0, 0, 0, 0, 0, 0}, lsq[8] = {0, 0, 0, 0, 0, 0, 0, 0};

  const int NT = 12;  // 12*16 = 192 >= 178 outputs
  for (int job = wv; job < CCH * NT; job += 8) {
    int c = job / NT, t0 = (job % NT) * 16;
    v8f acc;
#pragma unroll
    for (int v = 0; v < 8; ++v) acc[v] = b3r[v];
    for (int q = 0; q < 25; ++q) {  // K = 800 = 25 x 32
      int k0 = 32 * q;
      v16h af, bf;
#pragma unroll
      for (int r = 0; r < 8; ++r) {
        int kb = a_kbase(r, hi);
        af[2 * r]     = aW[ln][k0 + kb];
        af[2 * r + 1] = aW[ln][k0 + kb + 1];
      }
#pragma unroll
      for (int h = 0; h < 16; ++h) {
        int kidx = k0 + 16 * hi + h;       // < 800
        int g = kidx / KW, kk = kidx % KW; // input channel, tap
        int t = t0 + ln + kk;
        bf[h] = (t < PG1) ? in3[c][g][t] : (_Float16)0.f;
      }
      acc = __builtin_amdgcn_wmma_f32_16x16x32_f16(false, af, false, bf, (short)0, acc, false, false);
    }
    int tO = t0 + ln;
    if (tO < TO2) {
      int inPool = (tO < PG2 * 13);
      int g = tO / 13;
#pragma unroll
      for (int v = 0; v < 8; ++v) {
        float r = fmaxf(acc[v], 0.f);
        lsum[v] += r; lsq[v] += r * r;
        if (inPool) atomicMax(&pmax[c][v + 8 * hi][g], __float_as_int(r));
      }
    }
  }
#pragma unroll
  for (int v = 0; v < 8; ++v) {
    atomicAdd(&ssum[v + 8 * hi], lsum[v]);
    atomicAdd(&ssq[v + 8 * hi], lsq[v]);
  }
  __syncthreads();
  if (tid < F1) {
    atomicAdd(&ws[OFF_S2SUM + (size_t)w * F1 + tid], ssum[tid]);
    atomicAdd(&ws[OFF_S2SQ  + (size_t)w * F1 + tid], ssq[tid]);
  }
  float* p2 = ws + OFF_P2 + ((size_t)(w * B_SZ + b)) * F1 * PG2 * CCH;
  for (int i = tid; i < F1 * PG2 * CCH; i += 256) {
    int c = i % CCH, g = (i / CCH) % PG2, f = i / (CCH * PG2);
    p2[((size_t)f * PG2 + g) * CCH + c] = __int_as_float(pmax[c][f][g]);
  }
}

// ================= K3: BN2-fold + flatten + linear 416->100 =================
__global__ __launch_bounds__(256) void k3_linear(
    const float* __restrict__ g2, const float* __restrict__ be2,
    const float* __restrict__ wl, const float* __restrict__ bl,
    float* __restrict__ ws) {
  __shared__ float z[B_SZ][FEATN];
  __shared__ float scl[F1], shf[F1];
  const int w = blockIdx.x, tid = threadIdx.x;
  if (tid < F1) {
    const float N2 = (float)(B_SZ * TO2 * CCH);
    float m = ws[OFF_S2SUM + (size_t)w * F1 + tid] / N2;
    float var = ws[OFF_S2SQ + (size_t)w * F1 + tid] / N2 - m * m;
    float s = g2[tid] * rsqrtf(var + EPSBN);
    scl[tid] = s; shf[tid] = be2[tid] - m * s;
  }
  __syncthreads();
  const float* p2 = ws + OFF_P2 + (size_t)w * B_SZ * F1 * PG2 * CCH;
  for (int i = tid; i < B_SZ * FEATN; i += 256) {
    int b = i / FEATN, j = i % FEATN;
    int c = j & 1, g = (j >> 1) % PG2, f = (j >> 1) / PG2;  // j = (f*13+g)*2+c
    z[b][j] = p2[(((size_t)b * F1 + f) * PG2 + g) * CCH + c] * scl[f] + shf[f];
  }
  __syncthreads();
  for (int o = tid; o < B_SZ * NH; o += 256) {
    int b = o / NH, h = o % NH;
    float acc = bl[h];
    const float* wr = wl + (size_t)h * FEATN;
    for (int j = 0; j < FEATN; ++j) acc += wr[j] * z[b][j];
    ws[OFF_FEAT + ((size_t)(w * B_SZ + b)) * NH + h] = acc;
  }
}

// ================= K4: GRU over the 16 xc features (sequential, single block) =================
__global__ __launch_bounds__(256) void k4_gru(
    const float* __restrict__ wih, const float* __restrict__ whh,
    const float* __restrict__ bih, const float* __restrict__ bhh,
    float* __restrict__ ws) {
  __shared__ float hb[2][B_SZ][NH];
  const int tid = threadIdx.x;
  for (int i = tid; i < B_SZ * NH; i += 256) hb[0][i / NH][i % NH] = 0.f;
  __syncthreads();
  for (int s = 0; s < NP; ++s) {
    int cur = s & 1;
    const float* xbase = ws + OFF_FEAT + ((size_t)(NP * NB + s)) * B_SZ * NH;
    for (int o = tid; o < B_SZ * NH; o += 256) {
      int b = o / NH, h = o % NH;
      const float* x = xbase + b * NH;
      const float* hp = hb[cur][b];
      float ir = bih[h], iz = bih[NH + h], in_ = bih[2 * NH + h];
      float hr = bhh[h], hz = bhh[NH + h], hn_ = bhh[2 * NH + h];
      const float* wr = wih + (size_t)h * NH;
      const float* wz = wih + (size_t)(NH + h) * NH;
      const float* wn = wih + (size_t)(2 * NH + h) * NH;
      const float* ur = whh + (size_t)h * NH;
      const float* uz = whh + (size_t)(NH + h) * NH;
      const float* un = whh + (size_t)(2 * NH + h) * NH;
      for (int k = 0; k < NH; ++k) {
        float xv = x[k], hv = hp[k];
        ir += wr[k] * xv; iz += wz[k] * xv; in_ += wn[k] * xv;
        hr += ur[k] * hv; hz += uz[k] * hv; hn_ += un[k] * hv;
      }
      float rg = 1.f / (1.f + __expf(-(ir + hr)));
      float zg = 1.f / (1.f + __expf(-(iz + hz)));
      float ng = tanhf(in_ + rg * hn_);
      hb[cur ^ 1][b][h] = (1.f - zg) * ng + zg * hp[h];
    }
    __syncthreads();
  }
  for (int i = tid; i < B_SZ * NH; i += 256) ws[OFF_HN + i] = hb[0][i / NH][i % NH];
}

// ================= K5: bilinear scoring out[b,p,s] =================
__global__ __launch_bounds__(256) void k5_bilinear(
    const float* __restrict__ bw, const float* __restrict__ ws, float* __restrict__ out) {
  int id = blockIdx.x * 256 + threadIdx.x;
  if (id >= B_SZ * NP * (NB + 1)) return;
  int b = id / (NP * (NB + 1));
  int r = id % (NP * (NB + 1));
  int p = r / (NB + 1), s = r % (NB + 1);
  const float* feats = ws + OFF_FEAT;
  const float* cand = (s == 0)
      ? feats + ((size_t)(NP * NB + NP + p)) * B_SZ * NH + (size_t)b * NH    // xp
      : feats + ((size_t)(p * NB + (s - 1))) * B_SZ * NH + (size_t)b * NH;   // xb
  const float* hnv = ws + OFF_HN + (size_t)b * NH;
  float acc = 0.f;
  for (int h = 0; h < NH; ++h) {
    const float* wr = bw + ((size_t)(p * NH + h)) * NH;
    float inner = 0.f;
    for (int c = 0; c < NH; ++c) inner += wr[c] * cand[c];
    acc += hnv[h] * inner;
  }
  out[id] = acc;  // (b,p,s) row-major == (16,16,11)
}

extern "C" void kernel_launch(void* const* d_in, const int* in_sizes, int n_in,
                              void* d_out, int out_size, void* d_ws, size_t ws_size,
                              hipStream_t stream) {
  (void)in_sizes; (void)n_in; (void)out_size;
  if (ws_size < WS_FLOATS * sizeof(float)) return;  // needs ~92 MiB scratch

  const float* Xc  = (const float*)d_in[0];
  const float* Xp  = (const float*)d_in[1];
  const float* Xb  = (const float*)d_in[2];
  const float* w1  = (const float*)d_in[3];
  const float* b1  = (const float*)d_in[4];
  const float* w2  = (const float*)d_in[5];
  const float* b2  = (const float*)d_in[6];
  const float* g1  = (const float*)d_in[7];
  const float* be1 = (const float*)d_in[8];
  const float* w3  = (const float*)d_in[9];
  const float* b3  = (const float*)d_in[10];
  const float* g2  = (const float*)d_in[11];
  const float* be2 = (const float*)d_in[12];
  const float* wl  = (const float*)d_in[13];
  const float* bl  = (const float*)d_in[14];
  const float* wih = (const float*)d_in[15];
  const float* whh = (const float*)d_in[16];
  const float* bih = (const float*)d_in[17];
  const float* bhh = (const float*)d_in[18];
  const float* bw  = (const float*)d_in[19];
  float* ws  = (float*)d_ws;
  float* out = (float*)d_out;

  // zero BN-stat accumulators only (pooled/feat buffers are fully overwritten each call)
  hipMemsetAsync(d_ws, 0, 4 * (size_t)NWIN * F1 * sizeof(float), stream);

  k1_conv2<<<NWIN * B_SZ, 256, 0, stream>>>(Xc, Xp, Xb, w1, b1, w2, b2, ws);
  k2_conv3<<<NWIN * B_SZ, 256, 0, stream>>>(w3, b3, g1, be1, ws);
  k3_linear<<<NWIN, 256, 0, stream>>>(g2, be2, wl, bl, ws);
  k4_gru<<<1, 256, 0, stream>>>(wih, whh, bih, bhh, ws);
  int tot = B_SZ * NP * (NB + 1);
  k5_bilinear<<<(tot + 255) / 256, 256, 0, stream>>>(bw, ws, out);
}
